// FSE_Module_79147657331158
// MI455X (gfx1250) — matrix operand order repaired and checked by reading, not compiled, not run on hardware
//
#include <hip/hip_runtime.h>
#include <hip/hip_bf16.h>
#include <stdint.h>

typedef __bf16 bf16;
typedef __bf16 v16bf __attribute__((ext_vector_type(16)));
typedef __bf16 v8bf  __attribute__((ext_vector_type(8)));
typedef float  v8f   __attribute__((ext_vector_type(8)));
typedef unsigned int u32x4 __attribute__((ext_vector_type(4)));
typedef int i32x4 __attribute__((ext_vector_type(4)));
typedef int i32x8 __attribute__((ext_vector_type(8)));

#if defined(__has_builtin)
#if __has_builtin(__builtin_amdgcn_tensor_load_to_lds) && __has_builtin(__builtin_amdgcn_s_wait_tensorcnt)
#define HAVE_TDM 1
#endif
#endif
#ifndef HAVE_TDM
#define HAVE_TDM 0
#endif

__device__ __forceinline__ float mishf(float v) {
  float sp = (v > 15.f) ? v : log1pf(__expf(v));
  return v * tanhf(sp);
}

#if HAVE_TDM
// 2D tile load via Tensor Data Mover: rows x rowElems bf16,
// global row stride strideElems (in elements). Issued once per wave that calls it.
__device__ __forceinline__ void tdm_load_2d(const void* src, void* ldsDst,
                                            unsigned rowElems, unsigned rows,
                                            unsigned strideElems) {
  unsigned long long ga = (unsigned long long)(uintptr_t)src;
  unsigned ldsOff = (unsigned)(uintptr_t)ldsDst;  // low 32 bits of flat LDS ptr == LDS offset
  u32x4 g0;
  g0[0] = 1u;                                        // count=1, user descriptor
  g0[1] = ldsOff;                                    // lds_addr (bytes)
  g0[2] = (unsigned)ga;                              // global_addr[31:0]
  g0[3] = (unsigned)((ga >> 32) & 0x1FFFFFFull) | (2u << 30);  // addr[56:32] | type=2
  i32x8 g1;
  g1[0] = (int)(1u << 16);                           // data_size = 2 bytes; mask=0
  g1[1] = (int)((rowElems & 0xFFFFu) << 16);         // tensor_dim0[15:0]
  g1[2] = (int)(((rowElems >> 16) & 0xFFFFu) | ((rows & 0xFFFFu) << 16)); // dim0 hi | dim1 lo
  g1[3] = (int)(((rows >> 16) & 0xFFFFu) | ((rowElems & 0xFFFFu) << 16)); // dim1 hi | tile_dim0
  g1[4] = (int)(rows & 0xFFFFu);                     // tile_dim1 | tile_dim2=0
  g1[5] = (int)strideElems;                          // tensor_dim0_stride lo32
  g1[6] = 0;
  g1[7] = 0;
  i32x4 z4 = {0, 0, 0, 0};
  i32x8 z8 = {0, 0, 0, 0, 0, 0, 0, 0};
  __builtin_amdgcn_tensor_load_to_lds(g0, g1, z4, z4, z8, 0);
}
#endif

// ---------------- prep kernels ----------------
// x: fp32 NCHW [8][64][256][256] -> bf16 NHWC [8][256*256][64]
__global__ __launch_bounds__(256) void prep_x_bf(const float* __restrict__ x,
                                                 bf16* __restrict__ xbf) {
  int idx = blockIdx.x * 256 + threadIdx.x;  // 8*64*65536 = 33554432
  int p = idx & 65535;
  int c = (idx >> 16) & 63;
  int n = idx >> 22;
  xbf[((size_t)n * 65536 + p) * 64 + c] = (bf16)x[idx];
}

// w1: OIHW (128,64,3,3) -> [co][tap][ci] bf16, K=576
__global__ __launch_bounds__(256) void prep_w1(const float* __restrict__ w,
                                               bf16* __restrict__ wb) {
  int idx = blockIdx.x * 256 + threadIdx.x;
  if (idx >= 128 * 576) return;
  int co = idx / 576, r = idx % 576;
  int t = r >> 6, ci = r & 63;
  wb[idx] = (bf16)w[(co * 64 + ci) * 9 + t];
}

// w2: OIHW (64,128,3,3) -> [co][tap][ci] bf16, K=1152
__global__ __launch_bounds__(256) void prep_w2(const float* __restrict__ w,
                                               bf16* __restrict__ wb) {
  int idx = blockIdx.x * 256 + threadIdx.x;
  if (idx >= 64 * 1152) return;
  int co = idx / 1152, r = idx % 1152;
  int t = r >> 7, ci = r & 127;
  wb[idx] = (bf16)w[(co * 128 + ci) * 9 + t];
}

// wh: (64,192,1,1) -> bf16 [co][192]
__global__ __launch_bounds__(256) void prep_wh(const float* __restrict__ w,
                                               bf16* __restrict__ wb) {
  int idx = blockIdx.x * 256 + threadIdx.x;
  if (idx >= 64 * 192) return;
  wb[idx] = (bf16)w[idx];
}

// ---------------- conv1: 3x3 64->128, BN+Mish, out bf16 NHWC ----------------
__global__ __launch_bounds__(256) void conv1_wmma(
    const bf16* __restrict__ xbf, const bf16* __restrict__ w1b,
    const float* __restrict__ bb, const float* __restrict__ gg,
    const float* __restrict__ beb, const float* __restrict__ mmb,
    const float* __restrict__ vvb, bf16* __restrict__ h1) {
  constexpr int TX = 128, CB = 32, CIN = 64, K = 576, PW = TX + 2;
  __shared__ __attribute__((aligned(32))) bf16 act[3 * PW * CIN];  // 49920 B
  __shared__ __attribute__((aligned(32))) bf16 wsl[CB * K];        // 36864 B

  int b = blockIdx.x;
  int cb = b & 3;
  int xt = (b >> 2) & 1;
  int y = (b >> 3) & 255;
  int n = b >> 11;
  int x0 = xt * TX;

  int tid = threadIdx.x;
  int lane = tid & 31;
  int wv = tid >> 5;
  int mrow = lane & 15;
  int hh = lane >> 4;

  const bf16* wsrc = w1b + (size_t)cb * CB * K;
#if HAVE_TDM
  if (wv == 0) tdm_load_2d(wsrc, wsl, K, CB, K);
#else
  for (int i = tid; i < CB * K; i += 256) wsl[i] = wsrc[i];
#endif

  for (int i = tid; i < 3 * PW * CIN; i += 256) {
    int c = i & (CIN - 1);
    int p = (i / CIN) % PW;
    int r = i / (CIN * PW);
    int yy = y + r - 1;
    int xx = x0 + p - 1;
    bf16 val = (bf16)0.0f;
    if ((unsigned)yy < 256u && (unsigned)xx < 256u)
      val = xbf[((size_t)n * 65536 + yy * 256 + xx) * 64 + c];
    act[i] = val;
  }
#if HAVE_TDM
  if (wv == 0) __builtin_amdgcn_s_wait_tensorcnt(0);
#endif
  __syncthreads();

  v8f acc0 = {0.f, 0.f, 0.f, 0.f, 0.f, 0.f, 0.f, 0.f};
  v8f acc1 = {0.f, 0.f, 0.f, 0.f, 0.f, 0.f, 0.f, 0.f};
  int px = wv * 16;

#pragma unroll
  for (int t = 0; t < 9; ++t) {
    int ky = t / 3, kx = t % 3;
    const bf16* arow = &act[(ky * PW + px + mrow + kx) * CIN];
#pragma unroll
    for (int kk = 0; kk < 2; ++kk) {
      union { v16bf v; v8bf q[2]; } a;
      int c0 = kk * 32 + hh * 16;
      a.q[0] = *(const v8bf*)(arow + c0);
      a.q[1] = *(const v8bf*)(arow + c0 + 8);
      int kr = t * 64 + kk * 32 + hh * 16;
      v16bf b0 = *(const v16bf*)(&wsl[mrow * K + kr]);
      v16bf b1 = *(const v16bf*)(&wsl[(16 + mrow) * K + kr]);
      acc0 = __builtin_amdgcn_wmma_f32_16x16x32_bf16(false, a.v, false, b0,
                                                     (short)0, acc0, false, false);
      acc1 = __builtin_amdgcn_wmma_f32_16x16x32_bf16(false, a.v, false, b1,
                                                     (short)0, acc1, false, false);
    }
  }

#pragma unroll
  for (int ns = 0; ns < 2; ++ns) {
    v8f acc = ns ? acc1 : acc0;
    int co = cb * CB + ns * 16 + mrow;
    float s = gg[co] * rsqrtf(vvb[co] + 1e-5f);
    float tt = (bb[co] - mmb[co]) * s + beb[co];
    int xb = x0 + px + hh * 8;
    size_t base = ((size_t)n * 65536 + y * 256 + xb) * 128 + co;
#pragma unroll
    for (int r = 0; r < 8; ++r)
      h1[base + (size_t)r * 128] = (bf16)mishf(acc[r] * s + tt);
  }
}

// ---------------- conv2: 3x3 128->64, BN+Mish+residual, out fp32 NHWC -------
__global__ __launch_bounds__(256) void conv2_wmma(
    const bf16* __restrict__ h1b, const bf16* __restrict__ w2b,
    const float* __restrict__ bb, const float* __restrict__ gg,
    const float* __restrict__ beb, const float* __restrict__ mmb,
    const float* __restrict__ vvb, const float* __restrict__ xres,
    float* __restrict__ hout) {
  constexpr int TX = 64, CB = 32, CIN = 128, K = 1152, PW = TX + 2;
  __shared__ __attribute__((aligned(32))) bf16 act[3 * PW * CIN];  // 50688 B
  __shared__ __attribute__((aligned(32))) bf16 wsl[CB * K];        // 73728 B

  int b = blockIdx.x;
  int cb = b & 1;
  int xt = (b >> 1) & 3;
  int y = (b >> 3) & 255;
  int n = b >> 11;
  int x0 = xt * TX;

  int tid = threadIdx.x;
  int lane = tid & 31;
  int wv = tid >> 5;
  int mrow = lane & 15;
  int hh = lane >> 4;

  const bf16* wsrc = w2b + (size_t)cb * CB * K;
#if HAVE_TDM
  if (wv == 0) tdm_load_2d(wsrc, wsl, K, CB, K);
#else
  for (int i = tid; i < CB * K; i += 256) wsl[i] = wsrc[i];
#endif

  for (int i = tid; i < 3 * PW * CIN; i += 256) {
    int c = i & (CIN - 1);
    int p = (i >> 7) % PW;
    int r = i / (CIN * PW);
    int yy = y + r - 1;
    int xx = x0 + p - 1;
    bf16 val = (bf16)0.0f;
    if ((unsigned)yy < 256u && (unsigned)xx < 256u)
      val = h1b[((size_t)n * 65536 + yy * 256 + xx) * 128 + c];
    act[i] = val;
  }
#if HAVE_TDM
  if (wv == 0) __builtin_amdgcn_s_wait_tensorcnt(0);
#endif
  __syncthreads();

  int msub = wv & 3, nsub = wv >> 2;
  int px = msub * 16;
  v8f acc = {0.f, 0.f, 0.f, 0.f, 0.f, 0.f, 0.f, 0.f};

#pragma unroll
  for (int t = 0; t < 9; ++t) {
    int ky = t / 3, kx = t % 3;
    const bf16* arow = &act[(ky * PW + px + mrow + kx) * CIN];
#pragma unroll
    for (int kk = 0; kk < 4; ++kk) {
      union { v16bf v; v8bf q[2]; } a;
      int c0 = kk * 32 + hh * 16;
      a.q[0] = *(const v8bf*)(arow + c0);
      a.q[1] = *(const v8bf*)(arow + c0 + 8);
      int kr = t * 128 + kk * 32 + hh * 16;
      v16bf bfrag = *(const v16bf*)(&wsl[(nsub * 16 + mrow) * K + kr]);
      acc = __builtin_amdgcn_wmma_f32_16x16x32_bf16(false, a.v, false, bfrag,
                                                    (short)0, acc, false, false);
    }
  }

  int co = cb * CB + nsub * 16 + mrow;
  float s = gg[co] * rsqrtf(vvb[co] + 1e-5f);
  float tt = (bb[co] - mmb[co]) * s + beb[co];
  int xb = x0 + px + hh * 8;
  v8f res = *(const v8f*)(xres + ((size_t)(n * 64 + co)) * 65536 + y * 256 + xb);
  size_t base = ((size_t)n * 65536 + y * 256 + xb) * 64 + co;
#pragma unroll
  for (int r = 0; r < 8; ++r)
    hout[base + (size_t)r * 64] = mishf(acc[r] * s + tt) + res[r];
}

// -------- kernel 3: Haar DWT + 1x1 conv 192->64 (BN+Mish), writes d_out -----
__global__ __launch_bounds__(256) void dwt_hconv_wmma(
    const float* __restrict__ h, const bf16* __restrict__ whb,
    const float* __restrict__ bb, const float* __restrict__ gg,
    const float* __restrict__ beb, const float* __restrict__ mmb,
    const float* __restrict__ vvb, float* __restrict__ outLow,
    float* __restrict__ outHigh) {
  constexpr int PITCH = 200;  // 192 + pad, 400B rows (16B aligned)
  __shared__ __attribute__((aligned(32))) bf16 hi[64 * PITCH];  // 25600 B

  int b = blockIdx.x;
  int xt = b & 1;
  int yo = (b >> 1) & 127;
  int n = b >> 8;
  int x0 = xt * 64;

  int tid = threadIdx.x;
  int lane = tid & 31;
  int wv = tid >> 5;
  int mrow = lane & 15;
  int hh = lane >> 4;

  for (int i = tid; i < 64 * 64; i += 256) {
    int c = i & 63, xo = i >> 6;
    const float* hp =
        h + ((size_t)n * 65536 + (2 * yo) * 256 + 2 * (x0 + xo)) * 64 + c;
    float x00 = hp[0], x01 = hp[64], x10 = hp[64 * 256], x11 = hp[64 * 256 + 64];
    float cA = (x00 + x01 + x10 + x11) * 0.5f;
    float cH = (x00 + x01 - x10 - x11) * 0.5f;
    float cV = (x00 - x01 + x10 - x11) * 0.5f;
    float cD = (x00 - x01 - x10 + x11) * 0.5f;
    outLow[((size_t)n * 64 + c) * 16384 + yo * 128 + (x0 + xo)] = cA;
    hi[xo * PITCH + c] = (bf16)cH;
    hi[xo * PITCH + 64 + c] = (bf16)cV;
    hi[xo * PITCH + 128 + c] = (bf16)cD;
  }
  __syncthreads();

  int msub = wv & 3, npair = wv >> 2;
  v8f acc0 = {0.f, 0.f, 0.f, 0.f, 0.f, 0.f, 0.f, 0.f};
  v8f acc1 = {0.f, 0.f, 0.f, 0.f, 0.f, 0.f, 0.f, 0.f};

#pragma unroll
  for (int kk = 0; kk < 6; ++kk) {
    union { v16bf v; v8bf q[2]; } a;
    const bf16* arow = &hi[(msub * 16 + mrow) * PITCH];
    int c0 = kk * 32 + hh * 16;
    a.q[0] = *(const v8bf*)(arow + c0);
    a.q[1] = *(const v8bf*)(arow + c0 + 8);
    int kr = kk * 32 + hh * 16;
    v16bf b0 = *(const v16bf*)(whb + (size_t)(npair * 32 + mrow) * 192 + kr);
    v16bf b1 = *(const v16bf*)(whb + (size_t)(npair * 32 + 16 + mrow) * 192 + kr);
    acc0 = __builtin_amdgcn_wmma_f32_16x16x32_bf16(false, a.v, false, b0,
                                                   (short)0, acc0, false, false);
    acc1 = __builtin_amdgcn_wmma_f32_16x16x32_bf16(false, a.v, false, b1,
                                                   (short)0, acc1, false, false);
  }

#pragma unroll
  for (int ns = 0; ns < 2; ++ns) {
    v8f acc = ns ? acc1 : acc0;
    int co = npair * 32 + ns * 16 + mrow;
    float s = gg[co] * rsqrtf(vvb[co] + 1e-5f);
    float tt = (bb[co] - mmb[co]) * s + beb[co];
    v8f r8;
#pragma unroll
    for (int r = 0; r < 8; ++r) r8[r] = mishf(acc[r] * s + tt);
    int xb = x0 + msub * 16 + hh * 8;
    *(v8f*)(outHigh + ((size_t)n * 64 + co) * 16384 + yo * 128 + xb) = r8;
  }
}

// ---------------- launch ----------------
extern "C" void kernel_launch(void* const* d_in, const int* in_sizes, int n_in,
                              void* d_out, int out_size, void* d_ws,
                              size_t ws_size, hipStream_t stream) {
  (void)in_sizes; (void)n_in; (void)out_size; (void)ws_size;
  const float* x   = (const float*)d_in[0];
  const float* w1  = (const float*)d_in[1];
  const float* b1  = (const float*)d_in[2];
  const float* g1  = (const float*)d_in[3];
  const float* be1 = (const float*)d_in[4];
  const float* m1  = (const float*)d_in[5];
  const float* v1  = (const float*)d_in[6];
  const float* w2  = (const float*)d_in[7];
  const float* b2  = (const float*)d_in[8];
  const float* g2  = (const float*)d_in[9];
  const float* be2 = (const float*)d_in[10];
  const float* m2  = (const float*)d_in[11];
  const float* v2  = (const float*)d_in[12];
  const float* wh  = (const float*)d_in[13];
  const float* bh  = (const float*)d_in[14];
  const float* gh  = (const float*)d_in[15];
  const float* beh = (const float*)d_in[16];
  const float* mh  = (const float*)d_in[17];
  const float* vh  = (const float*)d_in[18];

  char* ws = (char*)d_ws;
  bf16*  xbf  = (bf16*)(ws + 0);           //  67,108,864 B
  bf16*  h1   = (bf16*)(ws + 67108864);    // 134,217,728 B
  float* hbuf = (float*)(ws + 201326592);  // 134,217,728 B
  bf16*  w1b  = (bf16*)(ws + 335544320);   //     147,456 B
  bf16*  w2b  = (bf16*)(ws + 335691776);   //     147,456 B
  bf16*  whb  = (bf16*)(ws + 335839232);   //      24,576 B

  float* outLow  = (float*)d_out;
  float* outHigh = outLow + (size_t)8 * 64 * 128 * 128;

  prep_x_bf<<<131072, 256, 0, stream>>>(x, xbf);
  prep_w1<<<(128 * 576 + 255) / 256, 256, 0, stream>>>(w1, w1b);
  prep_w2<<<(64 * 1152 + 255) / 256, 256, 0, stream>>>(w2, w2b);
  prep_wh<<<(64 * 192 + 255) / 256, 256, 0, stream>>>(wh, whb);
  conv1_wmma<<<16384, 256, 0, stream>>>(xbf, w1b, b1, g1, be1, m1, v1, h1);
  conv2_wmma<<<16384, 256, 0, stream>>>(h1, w2b, b2, g2, be2, m2, v2, x, hbuf);
  dwt_hconv_wmma<<<2048, 256, 0, stream>>>(hbuf, whb, bh, gh, beh, mh, vh,
                                           outLow, outHigh);
}
